// RNN_many_to_one_33328946217190
// MI455X (gfx1250) — matrix operand order repaired and checked by reading, not compiled, not run on hardware
//
#include <hip/hip_runtime.h>
#include <cstdint>
#include <cstddef>

#define B_ 64
#define S_ 2048
#define F_ 128
#define H_ 256
#define C_ 10

typedef __attribute__((ext_vector_type(16))) __bf16        bf16x16;
typedef __attribute__((ext_vector_type(8)))  float         floatx8;
typedef __attribute__((ext_vector_type(4)))  unsigned int  uintx4;

union Frag16 {
    bf16x16 bf;
    uintx4  q[2];
    unsigned short u16[16];
};

__device__ __forceinline__ unsigned short f32_to_bf16(float f) {
    unsigned int u = __builtin_bit_cast(unsigned int, f);
    u += 0x7FFFu + ((u >> 16) & 1u);   // round-to-nearest-even
    return (unsigned short)(u >> 16);
}

// native CDNA5 transcendental tanh (single TRANS op on the scan critical path)
__device__ __forceinline__ float fast_tanh(float x) {
#if defined(__HIP_DEVICE_COMPILE__)
#if __has_builtin(__builtin_amdgcn_tanhf)
    return __builtin_amdgcn_tanhf(x);
#else
    float r;
    asm("v_tanh_f32 %0, %1" : "=v"(r) : "v"(x));
    return r;
#endif
#else
    return tanhf(x);
#endif
}

// ---------------------------------------------------------------------------
// f32 -> bf16 conversion (grid-stride, streaming: non-temporal both ways)
// ---------------------------------------------------------------------------
__global__ void conv_f32_bf16(const float* __restrict__ src,
                              unsigned short* __restrict__ dst, int n) {
    for (int i = blockIdx.x * blockDim.x + threadIdx.x; i < n;
         i += gridDim.x * blockDim.x) {
        float v = __builtin_nontemporal_load(src + i);
        __builtin_nontemporal_store(f32_to_bf16(v), dst + i);
    }
}

// ---------------------------------------------------------------------------
// C[M,N] = A[M,K](bf16) @ W[N,K]^T(bf16) + b1[n] + b2[n]   (f32 out)
// one 16x16 output tile per wave, K-loop of v_wmma_f32_16x16x32_bf16
// ---------------------------------------------------------------------------
__global__ void gemm_bf16_wmma(const unsigned short* __restrict__ A,
                               const unsigned short* __restrict__ W,
                               const float* __restrict__ b1,
                               const float* __restrict__ b2,
                               float* __restrict__ Cout,
                               int M, int N, int K) {
    const int wave   = threadIdx.x >> 5;
    const int lane   = threadIdx.x & 31;
    const int gid    = blockIdx.x * (blockDim.x >> 5) + wave;
    const int tilesN = N >> 4;
    const int tileM  = gid / tilesN;
    const int tileN  = gid % tilesN;
    if (tileM * 16 >= M) return;

    const int rc    = lane & 15;   // row (A) / col (B) / col (C)
    const int khalf = lane >> 4;

    const unsigned short* Abase = A + (size_t)(tileM * 16 + rc) * K;
    const unsigned short* Wbase = W + (size_t)(tileN * 16 + rc) * K;

    floatx8 acc = {};
    const int nk = K >> 5;
    for (int kc = 0; kc < nk; ++kc) {
        Frag16 a, b;
        const int aoff = kc * 32 + khalf * 16;   // A: 16 contiguous K per lane-half
        a.q[0] = *reinterpret_cast<const uintx4*>(Abase + aoff);
        a.q[1] = *reinterpret_cast<const uintx4*>(Abase + aoff + 8);
        const int boff = kc * 32 + khalf * 16;   // B: 16 contiguous K per lane-half
        b.q[0] = *reinterpret_cast<const uintx4*>(Wbase + boff);
        b.q[1] = *reinterpret_cast<const uintx4*>(Wbase + boff + 8);
        acc = __builtin_amdgcn_wmma_f32_16x16x32_bf16(
            false, a.bf, false, b.bf, (short)0, acc, false, false);
    }

    const int n = tileN * 16 + rc;
    const float bias = b1[n] + b2[n];
#pragma unroll
    for (int r = 0; r < 8; ++r) {
        const int m = tileM * 16 + r + (khalf << 3);
        __builtin_nontemporal_store(acc[r] + bias, &Cout[(size_t)m * N + n]);
    }
}

// ---------------------------------------------------------------------------
// Elman recurrence: h_t = tanh(xproj[:,t,:] + h_{t-1} @ Whh^T)
// grid = 4 workgroups (16 batch rows each), 512 threads = 16 waves.
// Whh fragments register-resident across the whole scan; h double-buffered
// in LDS (one barrier/step); xproj register-prefetched one step ahead so
// HBM latency overlaps the WMMA chain + barrier.
// ---------------------------------------------------------------------------
__global__ void __launch_bounds__(512)
rnn_scan(const float* __restrict__ xproj,
         const unsigned short* __restrict__ Whh,   // bf16, [H,H] row-major
         unsigned short* __restrict__ hseq,        // bf16 [B,S,H] or null
         float* __restrict__ hlast,                // f32  [B,H]   or null
         int write_seq, int write_last) {
    __shared__ alignas(16) unsigned short hbuf[2][16][264]; // padded rows

    const int tid   = threadIdx.x;
    const int wave  = tid >> 5;
    const int lane  = tid & 31;
    const int rc    = lane & 15;
    const int khalf = lane >> 4;
    const int b0    = blockIdx.x * 16;  // batch slice
    const int n0    = wave * 16;        // column tile
    const int n     = n0 + rc;

    // preload B = Whh^T fragments: B[k][n] = Whh[n][k] -> contiguous in k
    Frag16 bw[8];
    const unsigned short* Wbase = Whh + (size_t)n * H_;
#pragma unroll
    for (int kc = 0; kc < 8; ++kc) {
        const int boff = kc * 32 + khalf * 16;
        bw[kc].q[0] = *reinterpret_cast<const uintx4*>(Wbase + boff);
        bw[kc].q[1] = *reinterpret_cast<const uintx4*>(Wbase + boff + 8);
    }

    // per-r xproj base indices (row m = r + 8*khalf of the C tile)
    size_t xbase[8];
#pragma unroll
    for (int r = 0; r < 8; ++r) {
        const int m = r + (khalf << 3);
        xbase[r] = (size_t)(b0 + m) * S_ * H_ + n;   // + t*H_ each step
    }

    // zero both h buffers (h_{-1} = 0)
    for (int i = tid; i < 2 * 16 * 264; i += 512)
        (&hbuf[0][0][0])[i] = 0;

    // prefetch xproj for t = 0 (streams once -> non-temporal)
    float xv[8];
#pragma unroll
    for (int r = 0; r < 8; ++r)
        xv[r] = __builtin_nontemporal_load(&xproj[xbase[r]]);

    __syncthreads();

    for (int t = 0; t < S_; ++t) {
        const int cur = t & 1;
        const int prv = cur ^ 1;

        // issue next step's xproj loads first: latency hides behind WMMAs
        const int tn = (t + 1 < S_) ? t + 1 : t;
        float xn[8];
#pragma unroll
        for (int r = 0; r < 8; ++r)
            xn[r] = __builtin_nontemporal_load(&xproj[xbase[r] + (size_t)tn * H_]);

        floatx8 acc = {};
        const unsigned short* hrow = &hbuf[prv][rc][0];
#pragma unroll
        for (int kc = 0; kc < 8; ++kc) {
            Frag16 a;
            const int aoff = kc * 32 + khalf * 16;
            a.q[0] = *reinterpret_cast<const uintx4*>(hrow + aoff);
            a.q[1] = *reinterpret_cast<const uintx4*>(hrow + aoff + 8);
            acc = __builtin_amdgcn_wmma_f32_16x16x32_bf16(
                false, a.bf, false, bw[kc].bf, (short)0, acc, false, false);
        }

#pragma unroll
        for (int r = 0; r < 8; ++r) {
            const int m = r + (khalf << 3);
            const float v = fast_tanh(acc[r] + xv[r]);
            const unsigned short vb = f32_to_bf16(v);
            hbuf[cur][m][n] = vb;
            if (write_seq)
                __builtin_nontemporal_store(vb, &hseq[xbase[r] + (size_t)t * H_]);
            if (write_last && t == S_ - 1)
                hlast[(b0 + m) * H_ + n] = v;
            xv[r] = xn[r];
        }
        __syncthreads();
    }
}

// ---------------------------------------------------------------------------
// out[b,c] = hlast[b,:] . Wd[c,:] + bd[c]   (64x10, trivial)
// ---------------------------------------------------------------------------
__global__ void head_kernel(const float* __restrict__ hlast,
                            const float* __restrict__ Wd,
                            const float* __restrict__ bd,
                            float* __restrict__ out) {
    const int idx = blockIdx.x * blockDim.x + threadIdx.x;
    if (idx >= B_ * C_) return;
    const int b = idx / C_;
    const int c = idx % C_;
    float s = bd[c];
    for (int h = 0; h < H_; ++h)
        s += hlast[b * H_ + h] * Wd[c * H_ + h];
    out[idx] = s;
}

// ---------------------------------------------------------------------------
extern "C" void kernel_launch(void* const* d_in, const int* in_sizes, int n_in,
                              void* d_out, int out_size, void* d_ws, size_t ws_size,
                              hipStream_t stream) {
    const float* x    = (const float*)d_in[0];
    const float* Wih0 = (const float*)d_in[1];
    const float* Whh0 = (const float*)d_in[2];
    const float* bih0 = (const float*)d_in[3];
    const float* bhh0 = (const float*)d_in[4];
    const float* Wih1 = (const float*)d_in[5];
    const float* Whh1 = (const float*)d_in[6];
    const float* bih1 = (const float*)d_in[7];
    const float* bhh1 = (const float*)d_in[8];
    const float* Wd   = (const float*)d_in[9];
    const float* bd   = (const float*)d_in[10];

    char* ws = (char*)d_ws;
    size_t off = 0;
    unsigned short* wih0b = (unsigned short*)(ws + off); off += (size_t)H_ * F_ * 2;       // 64 KB
    unsigned short* whh0b = (unsigned short*)(ws + off); off += (size_t)H_ * H_ * 2;       // 128 KB
    unsigned short* wih1b = (unsigned short*)(ws + off); off += (size_t)H_ * H_ * 2;
    unsigned short* whh1b = (unsigned short*)(ws + off); off += (size_t)H_ * H_ * 2;
    unsigned short* xb    = (unsigned short*)(ws + off); off += (size_t)B_ * S_ * F_ * 2;  // 32 MB
    float*          xproj = (float*)(ws + off);          off += (size_t)B_ * S_ * H_ * 4;  // 128 MB
    unsigned short* h1b   = (unsigned short*)(ws + off); off += (size_t)B_ * S_ * H_ * 2;  // 64 MB
    float*          hlast = (float*)(ws + off);          off += (size_t)B_ * H_ * 4;       // 64 KB

    const int M = B_ * S_;  // 131072

    // 1) convert weights + input to bf16
    conv_f32_bf16<<<128,  256, 0, stream>>>(Wih0, wih0b, H_ * F_);
    conv_f32_bf16<<<256,  256, 0, stream>>>(Whh0, whh0b, H_ * H_);
    conv_f32_bf16<<<256,  256, 0, stream>>>(Wih1, wih1b, H_ * H_);
    conv_f32_bf16<<<256,  256, 0, stream>>>(Whh1, whh1b, H_ * H_);
    conv_f32_bf16<<<8192, 256, 0, stream>>>(x,    xb,    M * F_);

    // 2) layer 0: xproj0 = x @ Wih0^T + (bih0 + bhh0)
    gemm_bf16_wmma<<<(M / 16) * (H_ / 16) / 8, 256, 0, stream>>>(
        xb, wih0b, bih0, bhh0, xproj, M, H_, F_);

    // 3) layer 0 scan -> h1 (bf16, for next GEMM)
    rnn_scan<<<B_ / 16, 512, 0, stream>>>(xproj, whh0b, h1b, nullptr, 1, 0);

    // 4) layer 1: xproj1 = h1 @ Wih1^T + (bih1 + bhh1)
    gemm_bf16_wmma<<<(M / 16) * (H_ / 16) / 8, 256, 0, stream>>>(
        h1b, wih1b, bih1, bhh1, xproj, M, H_, H_);

    // 5) layer 1 scan -> last hidden state only
    rnn_scan<<<B_ / 16, 512, 0, stream>>>(xproj, whh1b, nullptr, hlast, 0, 1);

    // 6) classifier head
    head_kernel<<<(B_ * C_ + 255) / 256, 256, 0, stream>>>(hlast, Wd, bd, (float*)d_out);
}
